// LSTMAggregator_33698313404803
// MI455X (gfx1250) — matrix operand order repaired and checked by reading, not compiled, not run on hardware
//
#include <hip/hip_runtime.h>
#include <hip/hip_bf16.h>
#include <math.h>

// ---------------- problem constants (from reference) ----------------
constexpr int NN   = 50000;   // nodes
constexpr int KS   = 10;      // neighbors / time steps
constexpr int FF   = 128;     // input dim
constexpr int HH   = 64;      // hidden per direction
constexpr int G4   = 4 * HH;  // 256 gate columns (i,f,g,o)
constexpr int BM   = 32;      // nodes per block (2 x 16-row WMMA tiles)
constexpr int NTHR = 256;     // 8 wave32s

typedef __attribute__((ext_vector_type(16))) _Float16 v16h;
typedef __attribute__((ext_vector_type(8)))  float    v8f;
typedef __attribute__((ext_vector_type(8)))  _Float16 h8;
typedef __attribute__((ext_vector_type(4)))  _Float16 h4;

__device__ __forceinline__ float sigmoidf_(float x) {
  return 1.0f / (1.0f + __expf(-x));
}

// Build a 16-half WMMA A-fragment from two adjacent 8-half LDS runs:
// lane holds row M = lane&15; K = [base..base+15],
// base = 16*(lane>=16) (matches the contiguous B-fragment order).
__device__ __forceinline__ v16h load_a_frag(const _Float16* p) {
  h8 lo = *(const h8*)p;
  h8 hi = *(const h8*)(p + 8);
  v16h r;
#pragma unroll
  for (int i = 0; i < 8; ++i) { r[i] = lo[i]; r[i + 8] = hi[i]; }
  return r;
}

// B-fragment loaded straight from the global f32 weight matrix with cvt:
// lane holds column N = lane&15 (= weight row), 16 contiguous K values
// starting at koff (koff includes 16*(lane>=16)).  L2-resident, once/block.
__device__ __forceinline__ v16h load_b_frag_g(const float* __restrict__ W,
                                              int ldw, int row, int koff) {
  const float4* p = (const float4*)(W + row * ldw + koff);
  v16h r;
#pragma unroll
  for (int q = 0; q < 4; ++q) {
    float4 f = p[q];
    r[q * 4 + 0] = (_Float16)f.x;
    r[q * 4 + 1] = (_Float16)f.y;
    r[q * 4 + 2] = (_Float16)f.z;
    r[q * 4 + 3] = (_Float16)f.w;
  }
  return r;
}

__global__ __launch_bounds__(NTHR)
void lstm_agg_kernel(const int*   __restrict__ idx,     // [NN, KS]
                     const float* __restrict__ embed,   // [V, FF]
                     const float* __restrict__ Wih_f,   // [G4, FF]
                     const float* __restrict__ Whh_f,   // [G4, HH]
                     const float* __restrict__ bih_f,
                     const float* __restrict__ bhh_f,
                     const float* __restrict__ Wih_b,   // [G4, FF]
                     const float* __restrict__ bih_b,
                     const float* __restrict__ bhh_b,
                     float*       __restrict__ out)     // [NN, 2*HH]
{
  // Only 52 KB LDS -- weights live in VGPRs for the whole kernel.
  __shared__ _Float16 sX[BM * FF];     //  8 KB  current-step inputs (f16)
  __shared__ _Float16 sH[BM * HH];     //  4 KB  hidden state (f16 WMMA A)
  __shared__ float    sC[BM * HH];     //  8 KB  cell state (f32)
  __shared__ float    sG[4 * BM * HH]; // 32 KB  activated gate planes i,f,g,o

  const int tid   = threadIdx.x;
  const int lane  = tid & 31;
  const int wave  = tid >> 5;
  const int gate  = wave & 3;       // 0:i 1:f 2:g 3:o  (torch gate order)
  const int ng    = wave >> 2;      // node half: rows [ng*16, ng*16+16)
  const int laneN = lane & 15;      // WMMA N column / A row within tile
  const int laneKA = (lane >> 4) * 16;  // A-fragment K sub-offset
  const int laneKB = (lane >> 4) * 16;  // B-fragment K sub-offset
  const int laneM  = (lane >> 4) * 8;   // C/D M offset of vgpr r
  const int blockBase = blockIdx.x * BM;

  // ---- hoist ALL forward weight fragments + biases into registers ----
  v16h wih[4][4];                       // [kt][ct]  16 frags = 128 VGPRs
  v16h whh[2][4];                       // [kt][ct]   8 frags =  64 VGPRs
  float bsum[4];
#pragma unroll
  for (int ct = 0; ct < 4; ++ct) {
    const int row = gate * HH + ct * 16 + laneN;
#pragma unroll
    for (int kt = 0; kt < 4; ++kt)
      wih[kt][ct] = load_b_frag_g(Wih_f, FF, row, kt * 32 + laneKB);
#pragma unroll
    for (int kt = 0; kt < 2; ++kt)
      whh[kt][ct] = load_b_frag_g(Whh_f, HH, row, kt * 32 + laneKB);
    bsum[ct] = bih_f[row] + bhh_f[row];
  }

  for (int i = tid; i < BM * HH; i += NTHR) sC[i] = 0.0f;

  // gather mapping: thread -> (row, 16-float chunk)
  const int grow   = tid >> 3;          // 0..31
  const int gchunk = (tid & 7) * 16;    // feature offset
  int nodeg = blockBase + grow;
  if (nodeg >= NN) nodeg = NN - 1;      // clamp tail (stores are guarded)

  // state-update mapping: thread -> (node m, 8 contiguous hidden cols)
  const int um  = tid >> 3;
  const int uj0 = (tid & 7) * 8;
  const int unode = blockBase + um;

  __syncthreads();

  // ======================= forward direction: 10 steps =====================
  for (int t = 0; t < KS; ++t) {
    // ---- gather embed rows for this step, f32 -> f16 into sX ----
    {
      int v = idx[nodeg * KS + t];
      const float4* src = (const float4*)(embed + (long)v * FF + gchunk);
      _Float16* dst = sX + grow * FF + gchunk;
#pragma unroll
      for (int q = 0; q < 4; ++q) {
        float4 f = src[q];
        h4 h = { (_Float16)f.x, (_Float16)f.y, (_Float16)f.z, (_Float16)f.w };
        *(h4*)(dst + q * 4) = h;
      }
      if (t + 1 < KS) {   // prefetch next step's row (global_prefetch_b8)
        int vn = idx[nodeg * KS + t + 1];
        __builtin_prefetch(embed + (long)vn * FF + gchunk, 0, 3);
      }
    }
    __syncthreads();

    // ---- preload ALL A-fragments for this step, then run the WMMA chain ----
    v16h ax[4];
    const _Float16* aBase = sX + (ng * 16 + laneN) * FF;
#pragma unroll
    for (int kt = 0; kt < 4; ++kt)
      ax[kt] = load_a_frag(aBase + kt * 32 + laneKA);
    v16h ah[2];
    if (t > 0) {
      const _Float16* hBase = sH + (ng * 16 + laneN) * HH;
#pragma unroll
      for (int kt = 0; kt < 2; ++kt)
        ah[kt] = load_a_frag(hBase + kt * 32 + laneKA);
    }

    // gates = x_t @ Wih^T + (bih+bhh) [+ h @ Whh^T]; all operands in VGPRs.
    v8f acc[4];
#pragma unroll
    for (int ct = 0; ct < 4; ++ct) {
      float b = bsum[ct];
      acc[ct] = (v8f){ b, b, b, b, b, b, b, b };
    }
#pragma unroll
    for (int kt = 0; kt < 4; ++kt)                   // FF = 4 * 32
#pragma unroll
      for (int ct = 0; ct < 4; ++ct)
        acc[ct] = __builtin_amdgcn_wmma_f32_16x16x32_f16(
            false, ax[kt], false, wih[kt][ct], (short)0, acc[ct], false, false);
    if (t > 0) {                                     // h == 0 at t == 0
#pragma unroll
      for (int kt = 0; kt < 2; ++kt)                 // HH = 2 * 32
#pragma unroll
        for (int ct = 0; ct < 4; ++ct)
          acc[ct] = __builtin_amdgcn_wmma_f32_16x16x32_f16(
              false, ah[kt], false, whh[kt][ct], (short)0, acc[ct], false, false);
    }

    // ---- activate own gate, publish plane to LDS ----
    float* gplane = sG + gate * (BM * HH) + (ng * 16) * HH;
#pragma unroll
    for (int ct = 0; ct < 4; ++ct) {
#pragma unroll
      for (int r = 0; r < 8; ++r) {
        float x = acc[ct][r];
        float y = (gate == 2) ? tanhf(x) : sigmoidf_(x);
        gplane[(laneM + r) * HH + ct * 16 + laneN] = y;
      }
    }
    __syncthreads();

    // ---- LSTM cell update: c = f*c + i*g ; h = o*tanh(c) ----
    {
      const float* gi = sG + 0 * BM * HH + um * HH + uj0;
      const float* gf = sG + 1 * BM * HH + um * HH + uj0;
      const float* gg = sG + 2 * BM * HH + um * HH + uj0;
      const float* go = sG + 3 * BM * HH + um * HH + uj0;
      float*    cp = sC + um * HH + uj0;
      _Float16* hp = sH + um * HH + uj0;
#pragma unroll
      for (int q = 0; q < 8; ++q) {
        float c = gf[q] * cp[q] + gi[q] * gg[q];
        float h = go[q] * tanhf(c);
        cp[q] = c;
        hp[q] = (_Float16)h;
        if (t == KS - 1 && unode < NN)
          out[(long)unode * (2 * HH) + uj0 + q] = h;   // forward half
      }
    }
    __syncthreads();
  }

  // =============== backward direction: single step on x[:, K-1] ============
  // h0 = c0 = 0  =>  c = i*g, h = o*tanh(c); Whh_b never touches the result.
  {
    v8f acc[4];
#pragma unroll
    for (int ct = 0; ct < 4; ++ct) {
      const int row = gate * HH + ct * 16 + laneN;
#pragma unroll
      for (int kt = 0; kt < 4; ++kt)   // overwrite forward frags with Wih_b
        wih[kt][ct] = load_b_frag_g(Wih_b, FF, row, kt * 32 + laneKB);
      float b = bih_b[row] + bhh_b[row];
      acc[ct] = (v8f){ b, b, b, b, b, b, b, b };
    }
    v16h ax[4];
    const _Float16* aBase = sX + (ng * 16 + laneN) * FF;   // still x[:, K-1]
#pragma unroll
    for (int kt = 0; kt < 4; ++kt)
      ax[kt] = load_a_frag(aBase + kt * 32 + laneKA);
#pragma unroll
    for (int kt = 0; kt < 4; ++kt)
#pragma unroll
      for (int ct = 0; ct < 4; ++ct)
        acc[ct] = __builtin_amdgcn_wmma_f32_16x16x32_f16(
            false, ax[kt], false, wih[kt][ct], (short)0, acc[ct], false, false);

    float* gplane = sG + gate * (BM * HH) + (ng * 16) * HH;
#pragma unroll
    for (int ct = 0; ct < 4; ++ct) {
#pragma unroll
      for (int r = 0; r < 8; ++r) {
        float x = acc[ct][r];
        float y = (gate == 2) ? tanhf(x) : sigmoidf_(x);
        gplane[(laneM + r) * HH + ct * 16 + laneN] = y;
      }
    }
  }
  __syncthreads();

  {
    const float* gi = sG + 0 * BM * HH + um * HH + uj0;
    const float* gg = sG + 2 * BM * HH + um * HH + uj0;
    const float* go = sG + 3 * BM * HH + um * HH + uj0;
    if (unode < NN) {
#pragma unroll
      for (int q = 0; q < 8; ++q) {
        float c = gi[q] * gg[q];
        float h = go[q] * tanhf(c);
        out[(long)unode * (2 * HH) + HH + uj0 + q] = h;  // backward half
      }
    }
  }
}

extern "C" void kernel_launch(void* const* d_in, const int* in_sizes, int n_in,
                              void* d_out, int out_size, void* d_ws, size_t ws_size,
                              hipStream_t stream) {
  (void)in_sizes; (void)n_in; (void)out_size; (void)d_ws; (void)ws_size;
  const int*   idx   = (const int*)  d_in[0];  // neigh_idx [N,K] int32
  const float* embed = (const float*)d_in[1];  // [V,F]
  const float* Wih_f = (const float*)d_in[2];
  const float* Whh_f = (const float*)d_in[3];
  const float* bih_f = (const float*)d_in[4];
  const float* bhh_f = (const float*)d_in[5];
  const float* Wih_b = (const float*)d_in[6];
  // d_in[7] = Whh_b is provably unused (backward output is step-0 hidden, h0=0)
  const float* bih_b = (const float*)d_in[8];
  const float* bhh_b = (const float*)d_in[9];
  float* out = (float*)d_out;

  const int grid = (NN + BM - 1) / BM;  // 1563 blocks of 256 threads
  lstm_agg_kernel<<<dim3(grid), dim3(NTHR), 0, stream>>>(
      idx, embed, Wih_f, Whh_f, bih_f, bhh_f, Wih_b, bih_b, bhh_b, out);
}
